// BasicFlow_60138132079052
// MI455X (gfx1250) — matrix operand order repaired and checked by reading, not compiled, not run on hardware
//
#include <hip/hip_runtime.h>
#include <math.h>

// ---------------------------------------------------------------------------
// BasicFlow forward for MI455X (gfx1250, wave32, WMMA).
// Pipeline: NHWC-bf16 prep -> implicit-GEMM conv3x3 (v_wmma_f32_16x16x32_bf16,
//           single-shot LDS halo staging via GLOBAL_LOAD_ASYNC_TO_LDS)
//           -> windowed QK^T correlation (WMMA) + softmax flows -> splice+upsample.
// ---------------------------------------------------------------------------

#define DIMC 128
#define HH   96
#define WWD  96
#define BN   8
#define NWIN 144          // 12*12 windows of 8x8
#define SCALE_F 0.08838834764831845f   // 128^-0.5 (folded into q)

// Async global->LDS staging (ASYNCcnt path), guarded so the build can't regress.
#if defined(__HIP_DEVICE_COMPILE__) && \
    __has_builtin(__builtin_amdgcn_global_load_async_to_lds_b128) && \
    __has_builtin(__builtin_amdgcn_s_wait_asynccnt)
#define USE_ASYNC_LDS 1
#else
#define USE_ASYNC_LDS 0
#endif

typedef __attribute__((ext_vector_type(16))) __bf16 v16bf;
typedef __attribute__((ext_vector_type(8)))  float  v8f;

#if USE_ASYNC_LDS
typedef int v4i_vs __attribute__((__vector_size__(16)));
typedef __attribute__((address_space(1))) v4i_vs* gv4i_p;   // global src
typedef __attribute__((address_space(3))) v4i_vs* lv4i_p;   // LDS dst
#endif

union FragBF { v16bf v; uint4 q[2]; };
union AccF   { v8f  v; float f[8]; };

__device__ __forceinline__ unsigned short f2bf(float x) {
  union { float f; unsigned u; } c; c.f = x;
  unsigned u = c.u;
  u += 0x7fffu + ((u >> 16) & 1u);       // round-to-nearest-even
  return (unsigned short)(u >> 16);
}
__device__ __forceinline__ int imax(int a, int b) { return a > b ? a : b; }
__device__ __forceinline__ int imin(int a, int b) { return a < b ? a : b; }

// ---------------------------------------------------------------------------
// prep: NCHW f32 -> NHWC bf16 (both feats in one launch)
// ---------------------------------------------------------------------------
__global__ __launch_bounds__(256) void prep_feat(const float* __restrict__ f0,
                                                 const float* __restrict__ f2,
                                                 unsigned short* __restrict__ t0,
                                                 unsigned short* __restrict__ t2) {
  int o = blockIdx.x * 256 + threadIdx.x;
  const int per = BN * DIMC * HH * WWD;
  if (o >= 2 * per) return;
  const float* src = (o < per) ? f0 : f2;
  unsigned short* dst = (o < per) ? t0 : t2;
  int r = (o < per) ? o : o - per;            // r = ((b*H+y)*W+x)*128 + c
  int c = r & 127;
  int s = r >> 7;
  int x = s % WWD; s /= WWD;
  int y = s % HH;
  int b = s / HH;
  dst[r] = f2bf(src[((b * DIMC + c) * HH + y) * WWD + x]);
}

// prep: weights OIHW f32 -> [tap][co][ci] bf16
__global__ __launch_bounds__(256) void prep_w(const float* __restrict__ wq,
                                              const float* __restrict__ wk,
                                              unsigned short* __restrict__ W2q,
                                              unsigned short* __restrict__ W2k) {
  int o = blockIdx.x * 256 + threadIdx.x;
  const int per = 9 * DIMC * DIMC;
  if (o >= 2 * per) return;
  const float* src = (o < per) ? wq : wk;
  unsigned short* dst = (o < per) ? W2q : W2k;
  int r = (o < per) ? o : o - per;            // r = (tap*128 + co)*128 + ci
  int ci  = r & 127;
  int co  = (r >> 7) & 127;
  int tap = r >> 14;
  dst[r] = f2bf(src[(co * DIMC + ci) * 9 + tap]);
}

// prep: expand relative-position bias table to bias64[l][t]
__global__ __launch_bounds__(256) void prep_bias(const float* __restrict__ bt,
                                                 float* __restrict__ bias64) {
  int o = blockIdx.x * 256 + threadIdx.x;
  if (o >= 4096) return;
  int l = o >> 6, t = o & 63;
  int ly = l >> 3, lx = l & 7, ty = t >> 3, tx = t & 7;
  int idx = (ly - ty + 7) * 15 + (lx - tx + 7);
  bias64[o] = bt[idx];
}

// ---------------------------------------------------------------------------
// conv3x3 implicit GEMM: M = 16 x-positions of one row, N = 128 co, K = 9*128.
// Block = 128 threads (4 waves); wave w owns co [32w, 32w+32) (two N tiles).
// The full 3x18x128 input halo is staged into LDS ONCE (via
// GLOBAL_LOAD_ASYNC_TO_LDS when available), then all 36 tap/chunk WMMA steps
// run barrier-free.  gridDim = (6, 96, 32), z = b*4 + which (0=q0 1=k0 2=q2 3=k2).
// ---------------------------------------------------------------------------
__global__ __launch_bounds__(128) void conv3x3_wmma(
    const unsigned short* __restrict__ featT0, const unsigned short* __restrict__ featT2,
    const unsigned short* __restrict__ W2q,    const unsigned short* __restrict__ W2k,
    const float* __restrict__ bq,              const float* __restrict__ bk,
    unsigned short* __restrict__ q0, unsigned short* __restrict__ k0,
    unsigned short* __restrict__ q2, unsigned short* __restrict__ k2) {
  int x0 = blockIdx.x * 16;
  int y  = blockIdx.y;
  int bz = blockIdx.z;
  int b = bz >> 2, which = bz & 3;
  const unsigned short* in = (which < 2) ? featT0 : featT2;
  const unsigned short* W2 = ((which & 1) == 0) ? W2q : W2k;
  const float* bias        = ((which & 1) == 0) ? bq : bk;
  unsigned short* out = (which == 0) ? q0 : (which == 1) ? k0 : (which == 2) ? q2 : k2;
  float oscale = ((which & 1) == 0) ? SCALE_F : 1.0f;   // fold 1/sqrt(C) into q

  // 3 rows x 18 x-positions x 128 ci of bf16 = 13824 B
  __shared__ __align__(16) unsigned short ldsIn[3 * 18 * DIMC];

  int tid = threadIdx.x, lane = tid & 31, wv = tid >> 5;
  int l4 = lane >> 4, lm = lane & 15;
  int co0 = wv * 32;

  // ---- one-shot halo staging (16B chunks) ----
  const int NV = 3 * 18 * (DIMC / 8);   // 864 uint4 chunks
  for (int i = tid; i < NV; i += 128) {
    int row = i / (18 * 16);
    int rem = i % (18 * 16);
    int xp  = rem / 16;                 // 0..17  (x = x0 + xp - 1)
    int c8  = rem % 16;                 // uint4 index within 128 ci
    int ysrc = y + row - 1;
    int xsrc = x0 + xp - 1;
    unsigned short* lp = ldsIn + (row * 18 + xp) * DIMC + c8 * 8;
    if (ysrc >= 0 && ysrc < HH && xsrc >= 0 && xsrc < WWD) {
      const unsigned short* gp = in + ((b * HH + ysrc) * WWD + xsrc) * DIMC + c8 * 8;
#if USE_ASYNC_LDS
      __builtin_amdgcn_global_load_async_to_lds_b128((gv4i_p)gp, (lv4i_p)lp, 0, 0);
#else
      *(uint4*)lp = *(const uint4*)gp;
#endif
    } else {
      *(uint4*)lp = make_uint4(0u, 0u, 0u, 0u);
    }
  }
#if USE_ASYNC_LDS
  __builtin_amdgcn_s_wait_asynccnt(0);
#endif
  __syncthreads();

  AccF acc0, acc1;
  #pragma unroll
  for (int r = 0; r < 8; r++) { acc0.f[r] = 0.f; acc1.f[r] = 0.f; }

  // ---- barrier-free MAC loop: 9 taps x 4 ci-chunks ----
  for (int tap = 0; tap < 9; tap++) {
    int ky = tap / 3, kx = tap % 3;
    const unsigned short* wrow = W2 + (tap * DIMC + (co0 + lm)) * DIMC;
    __builtin_prefetch(wrow + DIMC * DIMC, 0, 3);   // next tap's weights
    #pragma unroll
    for (int kc = 0; kc < 4; kc++) {
      // A fragment (16-bit A 16x32, this code's contiguous k order): lane {M=lm}, K = l4*16 + [0,16)
      const unsigned short* ap = ldsIn + (ky * 18 + lm + kx) * DIMC + kc * 32;
      FragBF a;
      a.q[0] = *(const uint4*)(ap + l4 * 16);
      a.q[1] = *(const uint4*)(ap + l4 * 16 + 8);
      // B fragments (16-bit B 32x16, same contiguous k order as A): lane {N=lm}, K = l4*16 + [0,16)
      const unsigned short* wp = wrow + kc * 32 + l4 * 16;
      FragBF b0, b1;
      b0.q[0] = *(const uint4*)(wp);             b0.q[1] = *(const uint4*)(wp + 8);
      b1.q[0] = *(const uint4*)(wp + 16 * DIMC); b1.q[1] = *(const uint4*)(wp + 16 * DIMC + 8);
      acc0.v = __builtin_amdgcn_wmma_f32_16x16x32_bf16(false, a.v, false, b0.v, (short)0, acc0.v, false, false);
      acc1.v = __builtin_amdgcn_wmma_f32_16x16x32_bf16(false, a.v, false, b1.v, (short)0, acc1.v, false, false);
    }
  }

  // epilogue: D layout -> lane holds N=lm, VGPR r holds M = r + 8*l4
  #pragma unroll
  for (int nt = 0; nt < 2; nt++) {
    AccF& acc = nt ? acc1 : acc0;
    int co = co0 + nt * 16 + lm;
    float bv = bias[co];
    #pragma unroll
    for (int r = 0; r < 8; r++) {
      int m = r + 8 * l4;
      out[((b * HH + y) * WWD + (x0 + m)) * DIMC + co] = f2bf((acc.f[r] + bv) * oscale);
    }
  }
}

// ---------------------------------------------------------------------------
// Correlation + flow for one (window, batch, dir*4+shiftvariant).
// Block = 64 threads (2 waves). QK^T 64x64 via WMMA -> LDS -> softmax flows.
// ---------------------------------------------------------------------------
__global__ __launch_bounds__(64) void corr_flow(
    const unsigned short* __restrict__ q0, const unsigned short* __restrict__ k0,
    const unsigned short* __restrict__ q2, const unsigned short* __restrict__ k2,
    const float* __restrict__ bias64,
    float* __restrict__ fmid, float* __restrict__ fbsd) {
  int win = blockIdx.x, b = blockIdx.y, z = blockIdx.z;
  int dir = z >> 2, var = z & 3;
  int sh = (var & 2) ? 4 : 0, sw = (var & 1) ? 4 : 0;
  const unsigned short* Q = dir ? q2 : q0;   // dir0: q0.k2 ; dir1: q2.k0
  const unsigned short* K = dir ? k0 : k2;
  int wy = win / 12, wx = win % 12;

  __shared__ float corr[64][64];
  __shared__ float cm[81], fmx81[81], fmy81[81];

  int tid = threadIdx.x, lane = tid & 31, wv = tid >> 5;
  int l4 = lane >> 4, lm = lane & 15;

  AccF acc[2][4];
  #pragma unroll
  for (int i = 0; i < 2; i++)
    #pragma unroll
    for (int j = 0; j < 4; j++)
      #pragma unroll
      for (int r = 0; r < 8; r++) acc[i][j].f[r] = 0.f;

  #pragma unroll
  for (int kc = 0; kc < 4; kc++) {
    FragBF a[2];
    #pragma unroll
    for (int mt = 0; mt < 2; mt++) {
      int row = wv * 32 + mt * 16 + lm;                       // query position l
      int yy = (wy * 8 + (row >> 3) + sh) % HH;               // roll via wrap
      int xx = (wx * 8 + (row & 7) + sw) % WWD;
      const unsigned short* p = Q + (size_t)((b * HH + yy) * WWD + xx) * DIMC + kc * 32 + l4 * 16;
      a[mt].q[0] = *(const uint4*)p;
      a[mt].q[1] = *(const uint4*)(p + 8);
    }
    #pragma unroll
    for (int nt = 0; nt < 4; nt++) {
      int t = nt * 16 + lm;                                   // key position t
      int yy = (wy * 8 + (t >> 3) + sh) % HH;
      int xx = (wx * 8 + (t & 7) + sw) % WWD;
      const unsigned short* p = K + (size_t)((b * HH + yy) * WWD + xx) * DIMC + kc * 32 + l4 * 16;
      FragBF bfr;
      bfr.q[0] = *(const uint4*)p; bfr.q[1] = *(const uint4*)(p + 8);
      acc[0][nt].v = __builtin_amdgcn_wmma_f32_16x16x32_bf16(false, a[0].v, false, bfr.v, (short)0, acc[0][nt].v, false, false);
      acc[1][nt].v = __builtin_amdgcn_wmma_f32_16x16x32_bf16(false, a[1].v, false, bfr.v, (short)0, acc[1][nt].v, false, false);
    }
  }

  // corr (+bias, +Swin mask) -> LDS.  q already carries SCALE.
  #pragma unroll
  for (int mt = 0; mt < 2; mt++)
    #pragma unroll
    for (int nt = 0; nt < 4; nt++)
      #pragma unroll
      for (int r = 0; r < 8; r++) {
        int lrow = wv * 32 + mt * 16 + r + 8 * l4;
        int tcol = nt * 16 + lm;
        float v = acc[mt][nt].f[r] + bias64[lrow * 64 + tcol];
        if (var) {
          int ly = wy * 8 + (lrow >> 3), lx = wx * 8 + (lrow & 7);
          int ty = wy * 8 + (tcol >> 3), tx = wx * 8 + (tcol & 7);
          int cly = (var & 2) ? ((ly < 88) ? 0 : (ly < 92) ? 1 : 2) : 0;
          int cty = (var & 2) ? ((ty < 88) ? 0 : (ty < 92) ? 1 : 2) : 0;
          int clx = (var & 1) ? ((lx < 88) ? 0 : (lx < 92) ? 1 : 2) : 0;
          int ctx = (var & 1) ? ((tx < 88) ? 0 : (tx < 92) ? 1 : 2) : 0;
          if (cly != cty || clx != ctx) v -= 10000.0f;
        }
        corr[lrow][tcol] = v;
      }
  __syncthreads();

  // _flow_bsd: central 4x4 query rows, softmax over 64 keys
  if (tid < 16) {
    int iy = tid >> 2, ix = tid & 3, ly = iy + 2, lx = ix + 2, l = ly * 8 + lx;
    float mx = -1e30f;
    for (int t = 0; t < 64; t++) mx = fmaxf(mx, corr[l][t]);
    float s = 0.f, fx = 0.f, fy = 0.f;
    for (int t = 0; t < 64; t++) {
      float e = __expf(corr[l][t] - mx);
      s += e; fx += e * (float)((t & 7) - lx); fy += e * (float)((t >> 3) - ly);
    }
    float* o = fbsd + (size_t)((((dir * 4 + var) * BN + b) * NWIN + win) * 2) * 16;
    o[tid] = fx / s;
    o[16 + tid] = fy / s;
  }

  // _flow_mid phase A: 81 half-pixel positions, softmax over 64 shifted corr values
  for (int p = tid; p < 81; p += 64) {
    int u = p / 9, v9 = p % 9;
    float mx = -1e30f;
    for (int t = 0; t < 64; t++) {
      int h2 = t >> 3, w2 = t & 7, h0 = u + 3 - h2, w0 = v9 + 3 - w2;
      float c = (h0 >= 0 && h0 < 8 && w0 >= 0 && w0 < 8) ? corr[h0 * 8 + w0][t] : 0.f;
      mx = fmaxf(mx, c);
    }
    float s = 0.f, fx = 0.f, fy = 0.f, cw = 0.f;
    float bx = 1.5f + 0.5f * (float)v9, by = 1.5f + 0.5f * (float)u;
    for (int t = 0; t < 64; t++) {
      int h2 = t >> 3, w2 = t & 7, h0 = u + 3 - h2, w0 = v9 + 3 - w2;
      float c = (h0 >= 0 && h0 < 8 && w0 >= 0 && w0 < 8) ? corr[h0 * 8 + w0][t] : 0.f;
      float e = __expf(c - mx);
      s += e; fx += e * ((float)w2 - bx); fy += e * ((float)h2 - by); cw += e * c;
    }
    cm[p] = cw / s; fmx81[p] = fx / s; fmy81[p] = fy / s;
  }
  __syncthreads();

  // _flow_mid phase B: 4-way quadrant softmax, x2 flow scale
  {
    int yq = tid >> 3, xq = tid & 7;
    int i00 = yq * 9 + xq, i01 = i00 + 1, i10 = i00 + 9, i11 = i00 + 10;
    float c0 = cm[i00], c1 = cm[i01], c2v = cm[i10], c3 = cm[i11];
    float mx = fmaxf(fmaxf(c0, c1), fmaxf(c2v, c3));
    float e0 = __expf(c0 - mx), e1 = __expf(c1 - mx), e2 = __expf(c2v - mx), e3 = __expf(c3 - mx);
    float s = e0 + e1 + e2 + e3;
    float ox = (fmx81[i00] * e0 + fmx81[i01] * e1 + fmx81[i10] * e2 + fmx81[i11] * e3) * 2.f / s;
    float oy = (fmy81[i00] * e0 + fmy81[i01] * e1 + fmy81[i10] * e2 + fmy81[i11] * e3) * 2.f / s;
    float* o = fmid + (size_t)((((dir * 4 + var) * BN + b) * NWIN + win) * 2) * 64;
    o[tid] = ox; o[64 + tid] = oy;
  }
}

// ---------------------------------------------------------------------------
// Fused splice (+roll) and bilinear upsample into the 4 concatenated outputs.
// ---------------------------------------------------------------------------
__device__ __forceinline__ float samp_mid(const float* __restrict__ fmid,
                                          int dir, int b, int ch, int Y, int X) {
  int Yp = (Y + 188) % 192, Xp = (X + 188) % 192;   // roll by +4
  int wy = Yp >> 4, hh = Yp & 15, wx = Xp >> 4, ww = Xp & 15;
  int var = ((hh >> 3) << 1) | (ww >> 3);
  int iy = hh & 7, ix = ww & 7;
  int win = wy * 12 + wx;
  return fmid[(size_t)((((dir * 4 + var) * BN + b) * NWIN + win) * 2 + ch) * 64 + iy * 8 + ix];
}
__device__ __forceinline__ float samp_bsd(const float* __restrict__ fbsd,
                                          int dir, int b, int ch, int Y, int X) {
  int Yp = (Y + 94) % 96, Xp = (X + 94) % 96;       // roll by +2
  int wy = Yp >> 3, hh = Yp & 7, wx = Xp >> 3, ww = Xp & 7;
  int var = ((hh >> 2) << 1) | (ww >> 2);
  int iy = hh & 3, ix = ww & 3;
  int win = wy * 12 + wx;
  return fbsd[(size_t)((((dir * 4 + var) * BN + b) * NWIN + win) * 2 + ch) * 16 + iy * 4 + ix];
}

__global__ __launch_bounds__(256) void splice_upsample(const float* __restrict__ fmid,
                                                       const float* __restrict__ fbsd,
                                                       float* __restrict__ out) {
  int gid = blockIdx.x * 256 + threadIdx.x;
  const int perT = BN * 2 * 384 * 384;
  if (gid >= 4 * perT) return;
  int ti = gid / perT;
  int r = gid % perT;
  int X = r % 384;
  int Y = (r / 384) % 384;
  int ch = (r / (384 * 384)) & 1;
  int b = r / (2 * 384 * 384);
  // output order: up2(flow10)=dir1 mid, up2(flow12)=dir0 mid, up4(flow02)=dir0 bsd, up4(flow20)=dir1 bsd
  int dir = (ti == 0 || ti == 3) ? 1 : 0;
  bool mid = (ti < 2);
  int S = mid ? 192 : 96;
  float f = mid ? 2.0f : 4.0f;
  float inv = mid ? 0.5f : 0.25f;
  float sy = (Y + 0.5f) * inv - 0.5f, sx = (X + 0.5f) * inv - 0.5f;
  int y0 = (int)floorf(sy), x0 = (int)floorf(sx);
  float wy = sy - (float)y0, wx = sx - (float)x0;
  int y0c = imax(y0, 0), y1c = imin(y0 + 1, S - 1);
  int x0c = imax(x0, 0), x1c = imin(x0 + 1, S - 1);
  float v00, v01, v10, v11;
  if (mid) {
    v00 = samp_mid(fmid, dir, b, ch, y0c, x0c); v01 = samp_mid(fmid, dir, b, ch, y0c, x1c);
    v10 = samp_mid(fmid, dir, b, ch, y1c, x0c); v11 = samp_mid(fmid, dir, b, ch, y1c, x1c);
  } else {
    v00 = samp_bsd(fbsd, dir, b, ch, y0c, x0c); v01 = samp_bsd(fbsd, dir, b, ch, y0c, x1c);
    v10 = samp_bsd(fbsd, dir, b, ch, y1c, x0c); v11 = samp_bsd(fbsd, dir, b, ch, y1c, x1c);
  }
  out[gid] = ((1.f - wy) * ((1.f - wx) * v00 + wx * v01) +
              wy * ((1.f - wx) * v10 + wx * v11)) * f;
}

// ---------------------------------------------------------------------------
extern "C" void kernel_launch(void* const* d_in, const int* in_sizes, int n_in,
                              void* d_out, int out_size, void* d_ws, size_t ws_size,
                              hipStream_t stream) {
  const float* feat0 = (const float*)d_in[0];
  const float* feat2 = (const float*)d_in[1];
  const float* wq = (const float*)d_in[2];
  const float* bq = (const float*)d_in[3];
  const float* wk = (const float*)d_in[4];
  const float* bk = (const float*)d_in[5];
  const float* bias_table = (const float*)d_in[6];
  float* out = (float*)d_out;
  (void)in_sizes; (void)n_in; (void)out_size; (void)ws_size;

  char* ws = (char*)d_ws;
  size_t off = 0;
  auto alloc = [&](size_t bytes) -> char* {
    char* p = ws + off;
    off = (off + bytes + 255) & ~(size_t)255;
    return p;
  };
  const size_t featElems = (size_t)BN * DIMC * HH * WWD;          // 9,437,184
  unsigned short* t0  = (unsigned short*)alloc(featElems * 2);
  unsigned short* t2  = (unsigned short*)alloc(featElems * 2);
  unsigned short* W2q = (unsigned short*)alloc((size_t)9 * DIMC * DIMC * 2);
  unsigned short* W2k = (unsigned short*)alloc((size_t)9 * DIMC * DIMC * 2);
  unsigned short* q0b = (unsigned short*)alloc(featElems * 2);
  unsigned short* k0b = (unsigned short*)alloc(featElems * 2);
  unsigned short* q2b = (unsigned short*)alloc(featElems * 2);
  unsigned short* k2b = (unsigned short*)alloc(featElems * 2);
  float* bias64 = (float*)alloc(4096 * 4);
  float* fmid   = (float*)alloc((size_t)2 * 4 * BN * NWIN * 2 * 64 * 4);
  float* fbsd   = (float*)alloc((size_t)2 * 4 * BN * NWIN * 2 * 16 * 4);

  { int total = 2 * (int)featElems;
    prep_feat<<<(total + 255) / 256, 256, 0, stream>>>(feat0, feat2, t0, t2); }
  { int total = 2 * 9 * DIMC * DIMC;
    prep_w<<<(total + 255) / 256, 256, 0, stream>>>(wq, wk, W2q, W2k); }
  prep_bias<<<16, 256, 0, stream>>>(bias_table, bias64);
  conv3x3_wmma<<<dim3(WWD / 16, HH, BN * 4), 128, 0, stream>>>(
      t0, t2, W2q, W2k, bq, bk, q0b, k0b, q2b, k2b);
  corr_flow<<<dim3(NWIN, BN, 8), 64, 0, stream>>>(q0b, k0b, q2b, k2b, bias64, fmid, fbsd);
  { int total = 4 * BN * 2 * 384 * 384;
    splice_upsample<<<(total + 255) / 256, 256, 0, stream>>>(fmid, fbsd, out); }
}
